// QuantWeightLeNet_80418967651022
// MI455X (gfx1250) — matrix operand order repaired and checked by reading, not compiled, not run on hardware
//
#include <hip/hip_runtime.h>
#include <stdint.h>

typedef __attribute__((ext_vector_type(16))) _Float16 v16h;
typedef __attribute__((ext_vector_type(8)))  _Float16 v8h;
typedef __attribute__((ext_vector_type(8)))  float    v8f;

// ---------------- workspace layout (weight region is B-independent) --------
static constexpr size_t WB1_OFF  = 0;        // 16 x 96  f16   conv1 B-matrix
static constexpr size_t WB2_OFF  = 8192;     // 16 x 160 f16   conv2 B-matrix
static constexpr size_t WFC1_OFF = 16384;    // 128 x 416 f16
static constexpr size_t WFC2_OFF = 131072;   // 96 x 128 f16
static constexpr size_t WFC3_OFF = 163840;   // 16 x 96  f16
static constexpr size_t BB1_OFF  = 172032;   // 16 f32
static constexpr size_t BB2_OFF  = 172160;   // 16 f32
static constexpr size_t BF1_OFF  = 172288;   // 128 f32
static constexpr size_t BF2_OFF  = 172928;   // 96 f32
static constexpr size_t ACT_BASE = 262144;   // activations start here

// ---------------- CDNA5 async global->LDS copies (ISA 08_async_tensor) -----
__device__ __forceinline__ void async_g2l_b128(uint32_t ldsaddr, uint32_t goff,
                                               const void* gbase) {
  asm volatile("global_load_async_to_lds_b128 %0, %1, %2"
               :: "v"(ldsaddr), "v"(goff), "s"(gbase) : "memory");
}
__device__ __forceinline__ void async_g2l_b32(uint32_t ldsaddr, uint32_t goff,
                                              const void* gbase) {
  asm volatile("global_load_async_to_lds_b32 %0, %1, %2"
               :: "v"(ldsaddr), "v"(goff), "s"(gbase) : "memory");
}
__device__ __forceinline__ void async_wait0() {
  asm volatile("s_wait_asynccnt 0" ::: "memory");
}

// ---------------- fragment loaders (16B-aligned contiguous halves) ---------
__device__ __forceinline__ v16h load_a16(const _Float16* base, int h) {
  // A 16x32 f16: elem 0..7 <- K = 8h+0..7 ; elem 8..15 <- K = 16+8h+0..7
  union { v16h v; v8h p[2]; } u;
  u.p[0] = *(const v8h*)(base + h * 8);
  u.p[1] = *(const v8h*)(base + 16 + h * 8);
  return u.v;
}
__device__ __forceinline__ v16h load_b16(const _Float16* base, int h) {
  // B 32x16 f16: lane holds column, elem 0..7 <- K = 8h+0..7 ; elem 8..15 <- K = 16+8h+0..7
  union { v16h v; v8h p[2]; } u;
  u.p[0] = *(const v8h*)(base + h * 8);
  u.p[1] = *(const v8h*)(base + 16 + h * 8);
  return u.v;
}

// ---------------- weight / bias quantization -------------------------------
__global__ __launch_bounds__(256) void quant_kernel(
    const float* w1, const float* b1, const float* w2, const float* b2,
    const float* wf1, const float* bf1, const float* wf2, const float* bf2,
    const float* wf3, char* ws)
{
  __shared__ float red[256];
  const int t = blockIdx.x, tid = threadIdx.x;
  if (t < 5) {
    const float* src; int N, inner, Npad, Kpad; _Float16* dst;
    switch (t) {
      case 0:  src = w1;  N = 6;   inner = 75;  Npad = 16;  Kpad = 96;
               dst = (_Float16*)(ws + WB1_OFF);  break;
      case 1:  src = w2;  N = 16;  inner = 150; Npad = 16;  Kpad = 160;
               dst = (_Float16*)(ws + WB2_OFF);  break;
      case 2:  src = wf1; N = 120; inner = 400; Npad = 128; Kpad = 416;
               dst = (_Float16*)(ws + WFC1_OFF); break;
      case 3:  src = wf2; N = 84;  inner = 120; Npad = 96;  Kpad = 128;
               dst = (_Float16*)(ws + WFC2_OFF); break;
      default: src = wf3; N = 10;  inner = 84;  Npad = 16;  Kpad = 96;
               dst = (_Float16*)(ws + WFC3_OFF); break;
    }
    for (int i = tid; i < Npad * Kpad; i += 256) dst[i] = (_Float16)0.f;
    float mx = 0.f;
    const int total = N * inner;
    for (int i = tid; i < total; i += 256) mx = fmaxf(mx, fabsf(src[i]));
    red[tid] = mx;
    __syncthreads();
    for (int s = 128; s > 0; s >>= 1) {
      if (tid < s) red[tid] = fmaxf(red[tid], red[tid + s]);
      __syncthreads();
    }
    const float scale = fmaxf(red[0] * (1.f / 3.f), 1e-20f);
    const float inv = 1.f / scale;
    for (int i = tid; i < total; i += 256) {
      float q = rintf(src[i] * inv);            // round-to-nearest-even
      q = fminf(fmaxf(q, -3.f), 3.f) * scale;   // clip to [-3,3] * scale
      dst[(i / inner) * Kpad + (i % inner)] = (_Float16)q;
    }
  } else {
    const float* src; int N, Npad; float* dst;
    switch (t) {
      case 5:  src = b1;  N = 6;   Npad = 16;  dst = (float*)(ws + BB1_OFF); break;
      case 6:  src = b2;  N = 16;  Npad = 16;  dst = (float*)(ws + BB2_OFF); break;
      case 7:  src = bf1; N = 120; Npad = 128; dst = (float*)(ws + BF1_OFF); break;
      default: src = bf2; N = 84;  Npad = 96;  dst = (float*)(ws + BF2_OFF); break;
    }
    for (int i = tid; i < Npad; i += 256) dst[i] = (i < N) ? src[i] : 0.f;
  }
}

// ---------------- conv1 + bias + relu + 2x2 maxpool (fused) ----------------
// One block = one image. Image (3x32x32 f32, 12KB) staged to LDS with async
// b128 copies; each of 8 waves computes two 8x8 conv-pixel blocks (4 WMMA
// M-tiles each, N-tile = 16 channels of which 6 used, K = 75 -> 96). All
// A-fragment gathers hit LDS, unconditional (clamped pixels / zero K-slot).
__global__ __launch_bounds__(256) void conv1_kernel(
    const float* __restrict__ x, char* __restrict__ ws, int B)
{
  __shared__ float lds_x[3073];                 // image + zero slot [3072]
  const int tid = threadIdx.x;
  const int lane = tid & 31, wave = tid >> 5;
  const int b = blockIdx.x;

  const float* gimg = x + (size_t)b * 3072;
  const uint32_t lbase = (uint32_t)(size_t)lds_x;
#pragma unroll
  for (int it = 0; it < 3; ++it) {
    const uint32_t off = (uint32_t)(tid + it * 256) * 16u;  // 768 x 16B chunks
    async_g2l_b128(lbase + off, off, gimg);
  }
  if (tid == 0) lds_x[3072] = 0.f;
  async_wait0();
  __syncthreads();

  const _Float16* Wb   = (const _Float16*)(ws + WB1_OFF);
  const float*    bias = (const float*)(ws + BB1_OFF);
  _Float16*       act1 = (_Float16*)(ws + ACT_BASE);

  const int h = lane >> 4, lm = lane & 15;
  const float bv = bias[lm];

  for (int half = 0; half < 2; ++half) {
    const int bl = wave + half * 8;
    const int BY = (bl >> 2) * 8, BX = (bl & 3) * 8;

    v8f d[4];
#pragma unroll
    for (int t = 0; t < 4; ++t)
#pragma unroll
      for (int v = 0; v < 8; ++v) d[t][v] = bv;

    int pixBase[4];
#pragma unroll
    for (int t = 0; t < 4; ++t) {
      const int m = t * 16 + lm;                // A row for this lane
      int oy = BY + (m >> 3), ox = BX + (m & 7);
      oy = (oy < 28) ? oy : 27;                 // clamp: results discarded
      ox = (ox < 28) ? ox : 27;
      pixBase[t] = oy * 32 + ox;
    }

#pragma unroll
    for (int kt = 0; kt < 3; ++kt) {
      const int kbase = kt * 32;
      const v16h bf = load_b16(Wb + lm * 96 + kbase, h);
#pragma unroll
      for (int t = 0; t < 4; ++t) {
        v16h af;
#pragma unroll
        for (int e = 0; e < 16; ++e) {
          const int K = kbase + ((e < 8) ? (e + 8 * h) : (8 + e + 8 * h));
          const int c = K / 25, r = K % 25;
          const int ofs = c * 1024 + (r / 5) * 32 + (r % 5);
          const int idx = (K < 75) ? (pixBase[t] + ofs) : 3072;
          af[e] = (_Float16)lds_x[idx];
        }
        d[t] = __builtin_amdgcn_wmma_f32_16x16x32_f16(
            false, af, false, bf, (short)0, d[t], false, false);
      }
    }

    // relu -> 2x2 maxpool: D row = v + 8h -> pixel (py = 2t+h, px = v)
    const int n = lm;
#pragma unroll
    for (int t = 0; t < 4; ++t) {
#pragma unroll
      for (int xp = 0; xp < 4; ++xp) {
        float p = fmaxf(fmaxf(d[t][2 * xp], 0.f), fmaxf(d[t][2 * xp + 1], 0.f));
        p = fmaxf(p, __shfl_xor(p, 16));        // combine py pair across halves
        const int PY = (BY >> 1) + t, PX = (BX >> 1) + xp;
        if (h == 0 && n < 6 && PY < 14 && PX < 14)
          act1[((b * 6 + n) * 14 + PY) * 14 + PX] = (_Float16)p;
      }
    }
  }
}

// ---------------- conv2 + bias + relu (implicit GEMM via LDS) --------------
// One block = one image. act1 image (6x14x14 f16, 2.3KB) staged async to LDS;
// 7 waves each compute one 16-pixel M-tile (100 pixels, padded to 112),
// N = 16 channels, K = 150 -> 160.
__global__ __launch_bounds__(256) void conv2_kernel(char* __restrict__ ws, int B)
{
  __shared__ _Float16 lds_a[1178];              // 1176 halfs + zero slot [1176]
  const int tid = threadIdx.x;
  const int lane = tid & 31, wave = tid >> 5;
  const int b = blockIdx.x;

  const _Float16* act1 = (const _Float16*)(ws + ACT_BASE);
  const _Float16* gin  = act1 + (size_t)b * 1176;
  const uint32_t lbase = (uint32_t)(size_t)lds_a;
#pragma unroll
  for (int it = 0; it < 3; ++it) {
    const int i = tid + it * 256;               // 588 x 4B chunks
    if (i < 588) async_g2l_b32(lbase + (uint32_t)i * 4u, (uint32_t)i * 4u, gin);
  }
  if (tid == 0) lds_a[1176] = (_Float16)0.f;
  async_wait0();
  __syncthreads();
  if (wave >= 7) return;

  const _Float16* Wb   = (const _Float16*)(ws + WB2_OFF);
  const float*    bias = (const float*)(ws + BB2_OFF);
  _Float16* act2 = (_Float16*)(ws + ACT_BASE + (size_t)B * 6 * 14 * 14 * 2);

  const int h = lane >> 4, lm = lane & 15;
  v8f d;
  const float bv = bias[lm];
#pragma unroll
  for (int v = 0; v < 8; ++v) d[v] = bv;

  // A-row pixel for this lane (clamped; clamped rows never stored)
  int m = wave * 16 + lm;
  m = (m < 100) ? m : 99;
  const int oy = m / 10, ox = m % 10;
  const int pixBase = oy * 14 + ox;

#pragma unroll
  for (int kt = 0; kt < 5; ++kt) {
    const int kbase = kt * 32;
    const v16h bf = load_b16(Wb + lm * 160 + kbase, h);
    v16h af;
#pragma unroll
    for (int e = 0; e < 16; ++e) {
      const int K = kbase + ((e < 8) ? (e + 8 * h) : (8 + e + 8 * h));
      const int c = K / 25, r = K % 25;
      const int ofs = c * 196 + (r / 5) * 14 + (r % 5);
      const int idx = (K < 150) ? (pixBase + ofs) : 1176;
      af[e] = lds_a[idx];
    }
    d = __builtin_amdgcn_wmma_f32_16x16x32_f16(
        false, af, false, bf, (short)0, d, false, false);
  }

  // store: D row = v + 8h -> pixel p (skip padded pixels)
#pragma unroll
  for (int v = 0; v < 8; ++v) {
    const int p = wave * 16 + v + 8 * h;
    if (p < 100) {
      const int py = p / 10, px = p % 10;
      act2[((b * 16 + lm) * 10 + py) * 10 + px] = (_Float16)fmaxf(d[v], 0.f);
    }
  }
}

// ---------------- 2x2 maxpool of conv2 -> padded flat [B,416] --------------
__global__ __launch_bounds__(256) void pool2_kernel(char* __restrict__ ws, int B)
{
  const int idx = blockIdx.x * 256 + threadIdx.x;   // over B*416
  if (idx >= B * 416) return;
  const _Float16* act2 =
      (const _Float16*)(ws + ACT_BASE + (size_t)B * 6 * 14 * 14 * 2);
  _Float16* act2p = (_Float16*)(ws + ACT_BASE + (size_t)B * 6 * 14 * 14 * 2 +
                                (size_t)B * 16 * 10 * 10 * 2);
  const int b = idx / 416, f = idx % 416;
  if (f < 400) {
    const int c = f / 25, r = f % 25, y = r / 5, xx = r % 5;
    const _Float16* in = act2 + ((b * 16 + c) * 10 + 2 * y) * 10 + 2 * xx;
    float m = fmaxf(fmaxf((float)in[0], (float)in[1]),
                    fmaxf((float)in[10], (float)in[11]));
    act2p[idx] = (_Float16)m;
  } else {
    act2p[idx] = (_Float16)0.f;   // zero K-pad for fc1
  }
}

// ---------------- generic fully-connected WMMA GEMM ------------------------
__global__ __launch_bounds__(256) void fc_kernel(
    const _Float16* __restrict__ X, const _Float16* __restrict__ W,
    const float* __restrict__ bias, void* __restrict__ Y,
    int Kpad, int ktiles, int ntiles, int totalTasks,
    int doRelu, int outF32, int outStride, int Nvalid)
{
  const int lane = threadIdx.x & 31, wave = threadIdx.x >> 5;
  const int task = blockIdx.x * 8 + wave;
  if (task >= totalTasks) return;
  const int mtile = task / ntiles, ntile = task % ntiles;
  const int h = lane >> 4, lm = lane & 15;
  const int ncol = ntile * 16 + lm;

  const float bv = bias ? bias[ncol] : 0.f;
  v8f c;
#pragma unroll
  for (int v = 0; v < 8; ++v) c[v] = bv;

  const _Float16* xrow = X + (size_t)(mtile * 16 + lm) * Kpad;
  const _Float16* wrow = W + (size_t)ncol * Kpad;
  for (int kt = 0; kt < ktiles; ++kt) {
    const int kbase = kt * 32;
    const v16h af = load_a16(xrow + kbase, h);
    const v16h bf = load_b16(wrow + kbase, h);
    c = __builtin_amdgcn_wmma_f32_16x16x32_f16(
        false, af, false, bf, (short)0, c, false, false);
  }

#pragma unroll
  for (int v = 0; v < 8; ++v) {
    const int row = mtile * 16 + v + 8 * h;
    float val = c[v];
    if (doRelu) val = fmaxf(val, 0.f);
    if (outF32) {
      if (ncol < Nvalid) ((float*)Y)[(size_t)row * outStride + ncol] = val;
    } else {
      ((_Float16*)Y)[(size_t)row * outStride + ncol] = (_Float16)val;
    }
  }
}

// ---------------- launcher -------------------------------------------------
extern "C" void kernel_launch(void* const* d_in, const int* in_sizes, int n_in,
                              void* d_out, int out_size, void* d_ws, size_t ws_size,
                              hipStream_t stream)
{
  const float* x   = (const float*)d_in[0];
  const float* w1  = (const float*)d_in[1];
  const float* b1  = (const float*)d_in[2];
  const float* w2  = (const float*)d_in[3];
  const float* b2  = (const float*)d_in[4];
  const float* wf1 = (const float*)d_in[5];
  const float* bf1 = (const float*)d_in[6];
  const float* wf2 = (const float*)d_in[7];
  const float* bf2 = (const float*)d_in[8];
  const float* wf3 = (const float*)d_in[9];
  char* ws = (char*)d_ws;
  const int B = in_sizes[0] / (3 * 32 * 32);   // 16384

  size_t off = ACT_BASE;
  off += (size_t)B * 6 * 14 * 14 * 2;          // act1   [B,6,14,14] f16
  off += (size_t)B * 16 * 10 * 10 * 2;         // act2   [B,16,10,10] f16
  _Float16* act2p = (_Float16*)(ws + off); off += (size_t)B * 416 * 2;
  _Float16* act3  = (_Float16*)(ws + off); off += (size_t)B * 128 * 2;
  _Float16* act4  = (_Float16*)(ws + off);

  const _Float16* Wfc1 = (const _Float16*)(ws + WFC1_OFF);
  const _Float16* Wfc2 = (const _Float16*)(ws + WFC2_OFF);
  const _Float16* Wfc3 = (const _Float16*)(ws + WFC3_OFF);
  const float* bf1p = (const float*)(ws + BF1_OFF);
  const float* bf2p = (const float*)(ws + BF2_OFF);

  quant_kernel<<<9, 256, 0, stream>>>(w1, b1, w2, b2, wf1, bf1, wf2, bf2, wf3, ws);

  conv1_kernel<<<B, 256, 0, stream>>>(x, ws, B);      // 1 image / block
  conv2_kernel<<<B, 256, 0, stream>>>(ws, B);         // 1 image / block
  pool2_kernel<<<(B * 416) / 256, 256, 0, stream>>>(ws, B);

  const int mt = B / 16;                              // 1024
  fc_kernel<<<(mt * 8) / 8, 256, 0, stream>>>(act2p, Wfc1, bf1p, act3,
      416, 13, 8, mt * 8, /*relu=*/1, /*f32=*/0, /*stride=*/128, 128);
  fc_kernel<<<(mt * 6) / 8, 256, 0, stream>>>(act3, Wfc2, bf2p, act4,
      128, 4, 6, mt * 6, /*relu=*/1, /*f32=*/0, /*stride=*/96, 96);
  fc_kernel<<<mt / 8, 256, 0, stream>>>(act4, Wfc3, nullptr, d_out,
      96, 3, 1, mt, /*relu=*/0, /*f32=*/1, /*stride=*/10, 10);
}
